// IrregularTimeAttentionLayer_48988396978196
// MI455X (gfx1250) — compile-verified, matrix operand order checked by reading, not run on hardware
//
#include <hip/hip_runtime.h>
#include <stdint.h>

// ---------------------------------------------------------------------------
// MI455X (gfx1250) attention layer. All GEMMs use v_wmma_f32_16x16x32_bf16
// (wave32, 16x16 f32 accumulators). B=8, S=1024, D=512, h=8, dk=64.
// All GEMM M/N dims are multiples of 128 and K is a multiple of 64, so the
// tiled kernel below needs no edge handling and can 2x-unroll the K loop.
// ---------------------------------------------------------------------------

typedef __bf16 bf16_t;
typedef __attribute__((ext_vector_type(4)))  float  v4f;
typedef __attribute__((ext_vector_type(8)))  float  v8f;
typedef __attribute__((ext_vector_type(8)))  bf16_t v8bf;
typedef __attribute__((ext_vector_type(16))) bf16_t v16bf;

// Explicit global address space: keeps GEMM operand traffic on
// global_load_b128 (LOADcnt only) instead of flat_load_b128 (LOADcnt+DScnt).
#define GAS __attribute__((address_space(1)))
typedef const GAS float  gcfloat;
typedef const GAS bf16_t gcbf16;
typedef GAS float        gfloat;
typedef GAS bf16_t       gbf16;

#define NB      8
#define SEQ     1024
#define DMODEL  512
#define NHEADS  8
#define DKH     64
#define HDIM    (NHEADS * DMODEL)   // 4096
#define LN_EPSF 1e-5f

static __device__ __forceinline__ v16bf bfcat(v8bf lo, v8bf hi)
{
    return __builtin_shufflevector(lo, hi, 0, 1, 2, 3, 4, 5, 6, 7,
                                   8, 9, 10, 11, 12, 13, 14, 15);
}

// One K-step (32 K-elements) worth of per-wave operand tiles:
// A: 2 M-subtiles, B: 4 N-subtiles.
template <int A_IS_F32>
struct KTiles {
    v4f  f[2][4];   // A staged as f32 (A_IS_F32 == 1)
    v8bf a[2][2];   // A staged as bf16 (A_IS_F32 == 0)
    v8bf b[4][2];
};

template <int A_IS_F32>
static __device__ __forceinline__ void load_tiles(
    KTiles<A_IS_F32>& T, gcfloat* const* pAf, gcbf16* const* pAb,
    gcbf16* const* pB, int koff)
{
#pragma unroll
    for (int mi = 0; mi < 2; ++mi) {
        if (A_IS_F32) {
            gcfloat* p = pAf[mi] + koff;
            T.f[mi][0] = *(const GAS v4f*)(p);
            T.f[mi][1] = *(const GAS v4f*)(p + 4);
            T.f[mi][2] = *(const GAS v4f*)(p + 16);
            T.f[mi][3] = *(const GAS v4f*)(p + 20);
        } else {
            gcbf16* p = pAb[mi] + koff;
            T.a[mi][0] = *(const GAS v8bf*)(p);
            T.a[mi][1] = *(const GAS v8bf*)(p + 16);
        }
    }
#pragma unroll
    for (int ni = 0; ni < 4; ++ni) {
        gcbf16* p = pB[ni] + koff;
        T.b[ni][0] = *(const GAS v8bf*)(p);
        T.b[ni][1] = *(const GAS v8bf*)(p + 16);
    }
}

template <int A_IS_F32>
static __device__ __forceinline__ void mma_tiles(const KTiles<A_IS_F32>& T,
                                                 v8f acc[2][4])
{
    v16bf amat[2];
#pragma unroll
    for (int mi = 0; mi < 2; ++mi) {
        if (A_IS_F32) {
            v16bf a;
#pragma unroll
            for (int i = 0; i < 4; ++i) {
                a[i]      = (bf16_t)T.f[mi][0][i];
                a[i + 4]  = (bf16_t)T.f[mi][1][i];
                a[i + 8]  = (bf16_t)T.f[mi][2][i];
                a[i + 12] = (bf16_t)T.f[mi][3][i];
            }
            amat[mi] = a;
        } else {
            amat[mi] = bfcat(T.a[mi][0], T.a[mi][1]);
        }
    }
#pragma unroll
    for (int ni = 0; ni < 4; ++ni) {
        const v16bf bmat = bfcat(T.b[ni][0], T.b[ni][1]);
#pragma unroll
        for (int mi = 0; mi < 2; ++mi) {
            acc[mi][ni] = __builtin_amdgcn_wmma_f32_16x16x32_bf16(
                false, amat[mi], false, bmat, (short)0, acc[mi][ni],
                false, false);
        }
    }
}

// ---------------------------------------------------------------------------
// WMMA GEMM:  C[m,n] = scale * sum_k A[m,k]*Bm[n,k]  (+ bias[n])
//   A : row-major [M,K]; f32 (converted in-register) or bf16  (template)
//   Bm: row-major [N,K] bf16 (computes A @ B^T -> contiguous K loads)
//   C : f32 or bf16 (template), leading dim ldc
// Batched over blockIdx.z: z = zb*zdiv + zh with per-component strides.
// Block = 256 threads = 8 waves arranged 4(M) x 2(N); each wave computes a
// 32x64 slab (2 M-subtiles x 4 N-subtiles = 8 accumulators) -> 128x128 block
// tile.  K-loop is 2x unrolled with ping-pong register buffers: each load
// stage issues a full K-step ahead of its consuming WMMAs.  K % 64 == 0.
// Per-lane operand layout follows CDNA5 ISA 7.12.2 (16-bit A 16x32): lanes
// 0-15 hold K{0..7,16..23}, lanes 16-31 hold K{8..15,24..31}.
// ---------------------------------------------------------------------------
template <int A_IS_F32, int C_IS_BF16>
__global__ __launch_bounds__(256) void gemm_wmma(
    const void* __restrict__ Av, const bf16_t* __restrict__ Bm,
    void* __restrict__ Cv, const float* __restrict__ bias,
    float scale, int K, int lda, int ldb, int ldc, int zdiv,
    long long sA1, long long sA2, long long sB1, long long sB2,
    long long sC1, long long sC2)
{
    const int z  = blockIdx.z;
    const int zb = z / zdiv;
    const int zh = z - zb * zdiv;

    gcfloat* Af = (gcfloat*)(uintptr_t)((const float*)Av  + zb * sA1 + zh * sA2);
    gcbf16*  Ab = (gcbf16*)(uintptr_t)((const bf16_t*)Av  + zb * sA1 + zh * sA2);
    gcbf16*  Bp = (gcbf16*)(uintptr_t)(Bm                 + zb * sB1 + zh * sB2);

    const int lane  = threadIdx.x & 31;
    const int wid   = threadIdx.x >> 5;
    const int wm    = wid >> 1;           // 0..3 (M)
    const int wn    = wid & 1;            // 0..1 (N)
    const int r16   = lane & 15;
    const int khalf = lane >> 4;          // which K-half this lane holds
    const int ko    = khalf * 8;

    const long long rowbase = (long long)blockIdx.y * 128 + wm * 32;
    const long long colbase = (long long)blockIdx.x * 128 + wn * 64;

    // per-lane operand pointers (advanced by 64 elements per unrolled step)
    gcfloat* pAf[2];
    gcbf16*  pAb[2];
#pragma unroll
    for (int mi = 0; mi < 2; ++mi) {
        const long long row = rowbase + mi * 16 + r16;
        if (A_IS_F32) pAf[mi] = Af + row * lda + ko;
        else          pAb[mi] = Ab + row * lda + ko;
    }
    gcbf16* pB[4];
#pragma unroll
    for (int ni = 0; ni < 4; ++ni)
        pB[ni] = Bp + (colbase + ni * 16 + r16) * (long long)ldb + ko;

    const v8f vzero = {};
    v8f acc[2][4];
#pragma unroll
    for (int mi = 0; mi < 2; ++mi)
#pragma unroll
        for (int ni = 0; ni < 4; ++ni) acc[mi][ni] = vzero;

    KTiles<A_IS_F32> T0, T1;

    // prologue: stage k = [0,32)
    load_tiles<A_IS_F32>(T0, pAf, pAb, pB, 0);

    // steady state: loads stay one K-step ahead of their WMMAs
    for (int k0 = 0; k0 + 64 < K; k0 += 64) {
        load_tiles<A_IS_F32>(T1, pAf, pAb, pB, 32);
        mma_tiles<A_IS_F32>(T0, acc);
#pragma unroll
        for (int mi = 0; mi < 2; ++mi) {
            if (A_IS_F32) pAf[mi] += 64;
            else          pAb[mi] += 64;
        }
#pragma unroll
        for (int ni = 0; ni < 4; ++ni) pB[ni] += 64;
        load_tiles<A_IS_F32>(T0, pAf, pAb, pB, 0);
        mma_tiles<A_IS_F32>(T1, acc);
    }

    // tail: last 64 K-elements (T0 already staged)
    load_tiles<A_IS_F32>(T1, pAf, pAb, pB, 32);
    mma_tiles<A_IS_F32>(T0, acc);
    mma_tiles<A_IS_F32>(T1, acc);

    // ---- store: C/D layout: VGPR r, lane l -> row = r + 8*(l>>4), col = l&15
    gfloat* Cf = (gfloat*)(uintptr_t)((float*)Cv  + zb * sC1 + zh * sC2);
    gbf16*  Cb = (gbf16*)(uintptr_t)((bf16_t*)Cv + zb * sC1 + zh * sC2);
    gcfloat* biasg = (gcfloat*)(uintptr_t)bias;
    const long long crowb = rowbase + 8 * khalf;

#pragma unroll
    for (int ni = 0; ni < 4; ++ni) {
        const long long ccol = colbase + ni * 16 + r16;
        const float bb = bias ? biasg[ccol] : 0.0f;
#pragma unroll
        for (int mi = 0; mi < 2; ++mi) {
#pragma unroll
            for (int r = 0; r < 8; ++r) {
                const float x = scale * acc[mi][ni][r] + bb;
                const long long ro = (crowb + mi * 16 + r) * (long long)ldc + ccol;
                if (C_IS_BF16) Cb[ro] = (bf16_t)x;
                else           Cf[ro] = x;
            }
        }
    }
}

// ---------------------------------------------------------------------------
// f32 -> bf16 elementwise convert (weights)
// ---------------------------------------------------------------------------
__global__ void f32_to_bf16_kernel(const float* __restrict__ src,
                                   bf16_t* __restrict__ dst, int n)
{
    int i = blockIdx.x * blockDim.x + threadIdx.x;
    if (i < n) dst[i] = (bf16_t)src[i];
}

// ---------------------------------------------------------------------------
// value [B,S,D] f32  ->  valueT [B,D,S] bf16 (LDS-tiled transpose, coalesced)
// grid (S/32, D/32, B), block (32, 8)
// ---------------------------------------------------------------------------
__global__ void transpose_value_bf16(const float* __restrict__ v,
                                     bf16_t* __restrict__ vT)
{
    __shared__ float tile[32][33];
    const int b  = blockIdx.z;
    const int s0 = blockIdx.x * 32;
    const int d0 = blockIdx.y * 32;
    const float* src = v  + (size_t)b * SEQ * DMODEL;
    bf16_t*      dst = vT + (size_t)b * DMODEL * SEQ;

#pragma unroll
    for (int j = 0; j < 4; ++j) {
        const int s = threadIdx.y + 8 * j;
        tile[s][threadIdx.x] = src[(size_t)(s0 + s) * DMODEL + d0 + threadIdx.x];
    }
    __syncthreads();
#pragma unroll
    for (int j = 0; j < 4; ++j) {
        const int d = threadIdx.y + 8 * j;
        dst[(size_t)(d0 + d) * SEQ + s0 + threadIdx.x] = (bf16_t)tile[threadIdx.x][d];
    }
}

// ---------------------------------------------------------------------------
// Row softmax, in place.  One block per row of SEQ=1024; 256 thr x 4 elems.
// ---------------------------------------------------------------------------
__global__ __launch_bounds__(256) void softmax_rows(float* __restrict__ attn)
{
    __shared__ float red[256];
    const int t = threadIdx.x;
    float* p = attn + (size_t)blockIdx.x * SEQ;

    float v[4];
    float m = -3.402823466e38f;
#pragma unroll
    for (int i = 0; i < 4; ++i) {
        v[i] = p[t + 256 * i];
        m = fmaxf(m, v[i]);
    }
    red[t] = m;
    __syncthreads();
    for (int o = 128; o > 0; o >>= 1) {
        if (t < o) red[t] = fmaxf(red[t], red[t + o]);
        __syncthreads();
    }
    m = red[0];
    __syncthreads();

    float s = 0.0f;
#pragma unroll
    for (int i = 0; i < 4; ++i) {
        v[i] = __expf(v[i] - m);
        s += v[i];
    }
    red[t] = s;
    __syncthreads();
    for (int o = 128; o > 0; o >>= 1) {
        if (t < o) red[t] += red[t + o];
        __syncthreads();
    }
    const float inv = 1.0f / red[0];

#pragma unroll
    for (int i = 0; i < 4; ++i) p[t + 256 * i] = v[i] * inv;
}

// ---------------------------------------------------------------------------
// Residual add + LayerNorm, in place over the projection output.
// One block per row of D=512; 256 threads x 2 elems.
// ---------------------------------------------------------------------------
__global__ __launch_bounds__(256) void residual_layernorm(
    float* __restrict__ outp, const float* __restrict__ value,
    const float* __restrict__ gamma, const float* __restrict__ beta)
{
    __shared__ float rs[256];
    __shared__ float rq[256];
    const int t = threadIdx.x;
    float*       orow = outp  + (size_t)blockIdx.x * DMODEL;
    const float* vrow = value + (size_t)blockIdx.x * DMODEL;

    const float y0 = orow[t]       + vrow[t];
    const float y1 = orow[t + 256] + vrow[t + 256];

    rs[t] = y0 + y1;
    rq[t] = y0 * y0 + y1 * y1;
    __syncthreads();
    for (int o = 128; o > 0; o >>= 1) {
        if (t < o) { rs[t] += rs[t + o]; rq[t] += rq[t + o]; }
        __syncthreads();
    }
    const float mean = rs[0] * (1.0f / DMODEL);
    const float var  = rq[0] * (1.0f / DMODEL) - mean * mean;
    const float rstd = rsqrtf(var + LN_EPSF);

    orow[t]       = (y0 - mean) * rstd * gamma[t]       + beta[t];
    orow[t + 256] = (y1 - mean) * rstd * gamma[t + 256] + beta[t + 256];
}

// ---------------------------------------------------------------------------
// Launch
// ---------------------------------------------------------------------------
extern "C" void kernel_launch(void* const* d_in, const int* in_sizes, int n_in,
                              void* d_out, int out_size, void* d_ws, size_t ws_size,
                              hipStream_t stream)
{
    (void)in_sizes; (void)n_in; (void)out_size; (void)ws_size;

    const float* query = (const float*)d_in[0];
    const float* key   = (const float*)d_in[1];
    const float* value = (const float*)d_in[2];
    const float* Wq    = (const float*)d_in[3];
    const float* bq    = (const float*)d_in[4];
    const float* Wk    = (const float*)d_in[5];
    const float* bk    = (const float*)d_in[6];
    const float* Wo    = (const float*)d_in[7];
    const float* bo    = (const float*)d_in[8];
    const float* gamma = (const float*)d_in[9];
    const float* beta  = (const float*)d_in[10];

    // d_out: [out (8*1024*512 f32)] ++ [attn (8*8*1024*1024 f32)]
    float* out_proj = (float*)d_out;
    float* attn     = (float*)d_out + (size_t)NB * SEQ * DMODEL;

    // workspace layout (bytes); total ~93 MB
    char* w = (char*)d_ws;
    bf16_t* wq_bf = (bf16_t*)(w + 0);          //  0.5 MB  Wq  [512,512]
    bf16_t* wk_bf = (bf16_t*)(w + 524288);     //  0.5 MB  Wk  [512,512]
    bf16_t* wo_bf = (bf16_t*)(w + 1048576);    //  4   MB  Wo  [512,4096]
    bf16_t* vT_bf = (bf16_t*)(w + 5242880);    //  8   MB  V^T [B,512,1024]
    bf16_t* q_bf  = (bf16_t*)(w + 13631488);   //  8   MB  Q   [8192,512]
    bf16_t* k_bf  = (bf16_t*)(w + 22020096);   //  8   MB  K   [8192,512]
    bf16_t* x_bf  = (bf16_t*)(w + 30408704);   // 64   MB  Xcat[8192,4096]

    // 1) weights -> bf16
    f32_to_bf16_kernel<<<(512 * 512 + 255) / 256, 256, 0, stream>>>(Wq, wq_bf, 512 * 512);
    f32_to_bf16_kernel<<<(512 * 512 + 255) / 256, 256, 0, stream>>>(Wk, wk_bf, 512 * 512);
    f32_to_bf16_kernel<<<(512 * 4096 + 255) / 256, 256, 0, stream>>>(Wo, wo_bf, 512 * 4096);

    // 2) value -> bf16 transposed [B, D, S]
    transpose_value_bf16<<<dim3(SEQ / 32, DMODEL / 32, NB), dim3(32, 8), 0, stream>>>(value, vT_bf);

    // 3) Q = query @ Wq^T + bq   (f32 A, bf16 out)   [8192,512]
    gemm_wmma<1, 1><<<dim3(DMODEL / 128, (NB * SEQ) / 128, 1), 256, 0, stream>>>(
        query, wq_bf, q_bf, bq, 1.0f,
        /*K=*/DMODEL, /*lda=*/DMODEL, /*ldb=*/DMODEL, /*ldc=*/DMODEL,
        1, 0, 0, 0, 0, 0, 0);

    // 4) K = key @ Wk^T + bk
    gemm_wmma<1, 1><<<dim3(DMODEL / 128, (NB * SEQ) / 128, 1), 256, 0, stream>>>(
        key, wk_bf, k_bf, bk, 1.0f,
        DMODEL, DMODEL, DMODEL, DMODEL,
        1, 0, 0, 0, 0, 0, 0);

    // 5) scores[b,h] = (1/8) * q[b,:,h*64:+64] @ k[b,:,h*64:+64]^T -> attn f32
    gemm_wmma<0, 0><<<dim3(SEQ / 128, SEQ / 128, NB * NHEADS), 256, 0, stream>>>(
        q_bf, k_bf, attn, nullptr, 0.125f,
        /*K=*/DKH, /*lda=*/DMODEL, /*ldb=*/DMODEL, /*ldc=*/SEQ,
        /*zdiv=*/NHEADS,
        /*sA1=*/(long long)SEQ * DMODEL, /*sA2=*/DKH,
        /*sB1=*/(long long)SEQ * DMODEL, /*sB2=*/DKH,
        /*sC1=*/(long long)NHEADS * SEQ * SEQ, /*sC2=*/(long long)SEQ * SEQ);

    // 6) softmax rows, in place (attn stays f32 in d_out, as reference output)
    softmax_rows<<<NB * NHEADS * SEQ, 256, 0, stream>>>(attn);

    // 7) xh[b,h] = attn[b,h] @ value[b]  (f32 A converted in-reg, bf16 out
    //    written into concat layout Xcat[b*S+s, h*512+d])
    gemm_wmma<1, 1><<<dim3(DMODEL / 128, SEQ / 128, NB * NHEADS), 256, 0, stream>>>(
        attn, vT_bf, x_bf, nullptr, 1.0f,
        /*K=*/SEQ, /*lda=*/SEQ, /*ldb=*/SEQ, /*ldc=*/HDIM,
        /*zdiv=*/NHEADS,
        /*sA1=*/(long long)NHEADS * SEQ * SEQ, /*sA2=*/(long long)SEQ * SEQ,
        /*sB1=*/(long long)DMODEL * SEQ, /*sB2=*/0,
        /*sC1=*/(long long)SEQ * HDIM, /*sC2=*/DMODEL);

    // 8) proj = Xcat @ Wo^T + bo  -> d_out[0 : B*S*D] f32
    gemm_wmma<0, 0><<<dim3(DMODEL / 128, (NB * SEQ) / 128, 1), 256, 0, stream>>>(
        x_bf, wo_bf, out_proj, bo, 1.0f,
        /*K=*/HDIM, /*lda=*/HDIM, /*ldb=*/HDIM, /*ldc=*/DMODEL,
        1, 0, 0, 0, 0, 0, 0);

    // 9) residual + LayerNorm, in place
    residual_layernorm<<<NB * SEQ, 256, 0, stream>>>(out_proj, value, gamma, beta);
}
